// TransformerModel_70411693851322
// MI455X (gfx1250) — matrix operand order repaired and checked by reading, not compiled, not run on hardware
//
#include <hip/hip_runtime.h>
#include <math.h>

// ---------------------------------------------------------------------------
// GPT-2-small forward pass for MI455X (gfx1250), bf16 WMMA everywhere.
// L=6 D=768 H=12 DK=64 FF=3072 V=50257 S=1024 B=2
// All GEMM B-operands stored [N,K] so per-lane gathers are contiguous b128.
// ---------------------------------------------------------------------------

#define LNUM   6
#define DMODEL 768
#define HNUM   12
#define DKV    64
#define FFDIM  3072
#define VOCAB  50257
#define SEQ    1024
#define BATCH  2
#define MTOK   (BATCH * SEQ)
#define EPSLN  1e-5f

typedef __bf16 bf16_t;
typedef __attribute__((ext_vector_type(2)))  __bf16 bf16x2;
typedef __attribute__((ext_vector_type(16))) __bf16 v16bf;
typedef __attribute__((ext_vector_type(8)))  float  v8f;

__device__ __forceinline__ v8f wmma_bf16(v16bf a, v16bf b, v8f c) {
  // D = A(16x32 bf16) * B(32x16 bf16) + C(16x16 f32)
  return __builtin_amdgcn_wmma_f32_16x16x32_bf16(false, a, false, b, (short)0, c,
                                                 false, false);
}

// A-matrix (16x32, 16-bit) K offset for VGPR pair r (contiguous order, k = 16*half + 2r).
__device__ __forceinline__ int a_koff(int r) { return 2 * r; }

// ---------------------------------------------------------------------------
// fp32 -> bf16 (no transpose) : embedding table (already [N,K] = [V,D])
// ---------------------------------------------------------------------------
__global__ void k_f32_to_bf16(const float* __restrict__ in, bf16_t* __restrict__ out,
                              long long n) {
  long long i = (long long)blockIdx.x * blockDim.x + threadIdx.x;
  long long stride = (long long)gridDim.x * blockDim.x;
  for (; i < n; i += stride) out[i] = (bf16_t)in[i];
}

// ---------------------------------------------------------------------------
// fp32 [K,N] -> bf16 [N,K] tiled transpose (weights)
// ---------------------------------------------------------------------------
__global__ __launch_bounds__(256) void k_cvt_transpose(const float* __restrict__ in,
                                                       bf16_t* __restrict__ out,
                                                       int K, int N) {
  __shared__ float tile[32][33];
  const int k0 = blockIdx.y * 32;
  const int n0 = blockIdx.x * 32;
  for (int j = threadIdx.y; j < 32; j += 8) {
    int kk = k0 + j, nn = n0 + threadIdx.x;
    tile[j][threadIdx.x] = (kk < K && nn < N) ? in[(size_t)kk * N + nn] : 0.f;
  }
  __syncthreads();
  for (int j = threadIdx.y; j < 32; j += 8) {
    int nn = n0 + j, kk = k0 + threadIdx.x;
    if (nn < N && kk < K) out[(size_t)nn * K + kk] = (bf16_t)tile[threadIdx.x][j];
  }
}

// ---------------------------------------------------------------------------
// x = emb[ids] + pe   (fp32 residual stream)
// ---------------------------------------------------------------------------
__global__ void k_embed(const int* __restrict__ ids, const float* __restrict__ emb,
                        const float* __restrict__ pe, float* __restrict__ xf) {
  int row = blockIdx.x;  // b*S + s
  int s = row % SEQ;
  int tok = ids[row];
  const float* e = emb + (size_t)tok * DMODEL;
  const float* p = pe + (size_t)s * DMODEL;
  float* o = xf + (size_t)row * DMODEL;
  for (int j = threadIdx.x; j < DMODEL; j += blockDim.x) o[j] = e[j] + p[j];
}

// ---------------------------------------------------------------------------
// LayerNorm fp32 row -> bf16 output
// ---------------------------------------------------------------------------
__global__ void k_layernorm_bf16(const float* __restrict__ x, const float* __restrict__ w,
                                 const float* __restrict__ b, bf16_t* __restrict__ out) {
  __shared__ float rs[256];
  __shared__ float rs2[256];
  int row = blockIdx.x;
  int tid = threadIdx.x;
  const float* xr = x + (size_t)row * DMODEL;
  float s = 0.f, s2 = 0.f;
  for (int j = tid; j < DMODEL; j += blockDim.x) { float v = xr[j]; s += v; s2 += v * v; }
  rs[tid] = s; rs2[tid] = s2;
  __syncthreads();
  for (int st = 128; st > 0; st >>= 1) {
    if (tid < st) { rs[tid] += rs[tid + st]; rs2[tid] += rs2[tid + st]; }
    __syncthreads();
  }
  float mu = rs[0] * (1.f / DMODEL);
  float var = rs2[0] * (1.f / DMODEL) - mu * mu;
  float rstd = rsqrtf(var + EPSLN);
  bf16_t* orow = out + (size_t)row * DMODEL;
  for (int j = tid; j < DMODEL; j += blockDim.x)
    orow[j] = (bf16_t)((xr[j] - mu) * rstd * w[j] + b[j]);
}

// ---------------------------------------------------------------------------
// WMMA GEMM: C[M,N] = A[M,K](bf16) * Bt[N,K](bf16) (+bias)(gelu)(+resid)
// One wave -> 16x64 tile (4 accumulators, A regs reused 4x).
// Block = 128 threads = 4 waves spanning 256 columns of N.
// OUT_VT scatters bf16 output as vt[B,H,DK,SEQ] (for attention V operand).
// ---------------------------------------------------------------------------
template <bool BIAS, bool RESID, bool GELU, bool OUTF, bool OUTB, bool OUT_VT>
__global__ __launch_bounds__(128) void k_gemm_wmma(
    const bf16_t* __restrict__ A, const bf16_t* __restrict__ Bt,
    const float* __restrict__ bias, const float* __restrict__ resid,
    float* __restrict__ Cf, bf16_t* __restrict__ Cb, int M, int N, int K) {
  const int lane = threadIdx.x & 31;
  const int wv = threadIdx.x >> 5;
  const int nbase = blockIdx.x * 256 + wv * 64;
  if (nbase >= N) return;
  const int mbase = blockIdx.y * 16;
  const int mcol = lane & 15;
  const int half = lane >> 4;
  const int kb = half * 16;  // A-layout K base for this lane half

  int nidx[4];
  bool val[4];
#pragma unroll
  for (int t = 0; t < 4; ++t) {
    int n = nbase + t * 16 + mcol;
    val[t] = (n < N);
    nidx[t] = val[t] ? n : (N - 1);
  }

  const bf16_t* Arow = A + (size_t)(mbase + mcol) * K + kb;
  v8f acc[4];
#pragma unroll
  for (int t = 0; t < 4; ++t)
#pragma unroll
    for (int i = 0; i < 8; ++i) acc[t][i] = 0.f;

  for (int kc = 0; kc < K; kc += 32) {
    v16bf a;
#pragma unroll
    for (int r = 0; r < 8; ++r) {
      bf16x2 u = *reinterpret_cast<const bf16x2*>(Arow + kc + a_koff(r));
      a[2 * r] = u[0]; a[2 * r + 1] = u[1];
    }
    if (kc + 32 < K) __builtin_prefetch(Arow + kc + 32, 0, 1);
    const int krow = kc + 16 * half;  // B-layout K base for this lane half
#pragma unroll
    for (int t = 0; t < 4; ++t) {
      const bf16_t* Bp = Bt + (size_t)nidx[t] * K + krow;
      v16bf b;
#pragma unroll
      for (int r = 0; r < 8; ++r) {
        bf16x2 u = *reinterpret_cast<const bf16x2*>(Bp + 2 * r);
        b[2 * r] = u[0]; b[2 * r + 1] = u[1];
      }
      acc[t] = wmma_bf16(a, b, acc[t]);
    }
  }

#pragma unroll
  for (int t = 0; t < 4; ++t) {
    const int n = nbase + t * 16 + mcol;
    const float bv = BIAS ? bias[nidx[t]] : 0.f;
#pragma unroll
    for (int i = 0; i < 8; ++i) {
      const int row = mbase + half * 8 + i;  // C layout: row = vgpr + 8*half
      float vv = acc[t][i] + bv;
      if (GELU) vv = 0.5f * vv * (1.f + erff(vv * 0.70710678118654752f));
      if (val[t]) {
        if (RESID) vv += resid[(size_t)row * N + n];
        if (OUTF) Cf[(size_t)row * N + n] = vv;
        if (OUTB) Cb[(size_t)row * N + n] = (bf16_t)vv;
        if (OUT_VT) {
          const int s = row % SEQ, bb = row / SEQ;
          const int h = n / DKV, dk = n % DKV;
          Cb[(((size_t)bb * HNUM + h) * DKV + dk) * SEQ + s] = (bf16_t)vv;
        }
      }
    }
  }
}

// ---------------------------------------------------------------------------
// Flash attention: one wave owns 16 query rows of one (b,h); key chunks of 32.
// scores: 4x WMMA, online softmax in C-layout, P restaged via LDS, PV: 4x WMMA.
// V consumed pre-transposed: vt[B,H,DK,SEQ] -> contiguous key gathers.
// ---------------------------------------------------------------------------
__global__ __launch_bounds__(128) void k_flash_attn(
    const bf16_t* __restrict__ q, const bf16_t* __restrict__ k,
    const bf16_t* __restrict__ vt, const int* __restrict__ am,
    bf16_t* __restrict__ o) {
  __shared__ float pst[4][16 * 32];
  const int lane = threadIdx.x & 31;
  const int wv = threadIdx.x >> 5;
  const int bh = blockIdx.x;
  const int b = bh / HNUM;
  const int h = bh % HNUM;
  const int qbase = (blockIdx.y * 4 + wv) * 16;
  const int mcol = lane & 15;
  const int half = lane >> 4;
  const int kb = half * 16;
  float* P = pst[wv];

  // Q tile (16x64) as two A-operands (d 0..31 / 32..63)
  v16bf aq0, aq1;
  const bf16_t* Qp = q + ((size_t)(b * SEQ + qbase + mcol)) * DMODEL + h * DKV;
#pragma unroll
  for (int r = 0; r < 8; ++r) {
    const int ko = kb + a_koff(r);
    bf16x2 t0 = *reinterpret_cast<const bf16x2*>(Qp + ko);
    bf16x2 t1 = *reinterpret_cast<const bf16x2*>(Qp + 32 + ko);
    aq0[2 * r] = t0[0]; aq0[2 * r + 1] = t0[1];
    aq1[2 * r] = t1[0]; aq1[2 * r + 1] = t1[1];
  }

  v8f m_run, l_run, ot[4];
#pragma unroll
  for (int i = 0; i < 8; ++i) {
    m_run[i] = -1e30f; l_run[i] = 0.f;
    ot[0][i] = ot[1][i] = ot[2][i] = ot[3][i] = 0.f;
  }

  const bf16_t* Vh = vt + ((size_t)b * HNUM + h) * DKV * SEQ;

  const int kend = qbase + 16;  // causal: keys < kend
  for (int kbase = 0; kbase < kend; kbase += 32) {
    v8f s0 = {0.f, 0.f, 0.f, 0.f, 0.f, 0.f, 0.f, 0.f};
    v8f s1 = s0;
    // score tiles: keys [kbase, kbase+16) and [kbase+16, kbase+32)
#pragma unroll
    for (int t = 0; t < 2; ++t) {
      const int key = kbase + t * 16 + mcol;  // B layout: lane -> column (key)
      const bf16_t* Kp = k + ((size_t)(b * SEQ + key)) * DMODEL + h * DKV + 16 * half;
      v16bf bk0, bk1;
#pragma unroll
      for (int r = 0; r < 8; ++r) {
        bf16x2 u0 = *reinterpret_cast<const bf16x2*>(Kp + 2 * r);
        bf16x2 u1 = *reinterpret_cast<const bf16x2*>(Kp + 32 + 2 * r);
        bk0[2 * r] = u0[0]; bk0[2 * r + 1] = u0[1];
        bk1[2 * r] = u1[0]; bk1[2 * r + 1] = u1[1];
      }
      if (t == 0) { s0 = wmma_bf16(aq0, bk0, s0); s0 = wmma_bf16(aq1, bk1, s0); }
      else        { s1 = wmma_bf16(aq0, bk0, s1); s1 = wmma_bf16(aq1, bk1, s1); }
    }

    // scale + causal/pad mask; C layout: element i -> row (half*8+i), col mcol(+16)
    const int key0 = kbase + mcol, key1 = key0 + 16;
    const bool keep0 = am[b * SEQ + key0] != 0;
    const bool keep1 = am[b * SEQ + key1] != 0;
    v8f cm;
#pragma unroll
    for (int i = 0; i < 8; ++i) {
      const int qrow = qbase + half * 8 + i;
      float a0 = (keep0 && key0 <= qrow) ? s0[i] * 0.125f : -1e9f;
      float a1 = (keep1 && key1 <= qrow) ? s1[i] * 0.125f : -1e9f;
      s0[i] = a0; s1[i] = a1; cm[i] = fmaxf(a0, a1);
    }
    // row max across the 16 lanes of this half (xor masks stay within half)
#pragma unroll
    for (int mm = 1; mm < 16; mm <<= 1)
#pragma unroll
      for (int i = 0; i < 8; ++i) cm[i] = fmaxf(cm[i], __shfl_xor(cm[i], mm, 32));

    v8f alpha, rsum;
#pragma unroll
    for (int i = 0; i < 8; ++i) {
      const float mn = fmaxf(m_run[i], cm[i]);
      alpha[i] = __expf(m_run[i] - mn);
      m_run[i] = mn;
      s0[i] = __expf(s0[i] - mn);
      s1[i] = __expf(s1[i] - mn);
      rsum[i] = s0[i] + s1[i];
    }
#pragma unroll
    for (int mm = 1; mm < 16; mm <<= 1)
#pragma unroll
      for (int i = 0; i < 8; ++i) rsum[i] += __shfl_xor(rsum[i], mm, 32);
#pragma unroll
    for (int i = 0; i < 8; ++i) {
      l_run[i] = l_run[i] * alpha[i] + rsum[i];
      ot[0][i] *= alpha[i]; ot[1][i] *= alpha[i];
      ot[2][i] *= alpha[i]; ot[3][i] *= alpha[i];
    }

    // restage P (C layout -> 16x32 tile in LDS -> A layout)
#pragma unroll
    for (int i = 0; i < 8; ++i) {
      P[(half * 8 + i) * 32 + mcol] = s0[i];
      P[(half * 8 + i) * 32 + 16 + mcol] = s1[i];
    }
    asm volatile("s_wait_dscnt 0" ::: "memory");
    v16bf pa;
#pragma unroll
    for (int r = 0; r < 8; ++r) {
      const int ko = kb + a_koff(r);
      pa[2 * r]     = (bf16_t)P[mcol * 32 + ko];
      pa[2 * r + 1] = (bf16_t)P[mcol * 32 + ko + 1];
    }
    asm volatile("s_wait_dscnt 0" ::: "memory");

    // O += P(16x32) * V(32x64) as 4 N-tiles of 16; vt rows contiguous in keys
#pragma unroll
    for (int t = 0; t < 4; ++t) {
      const int d = t * 16 + mcol;
      const bf16_t* Vp = Vh + (size_t)d * SEQ + kbase + 16 * half;
      v16bf bv;
#pragma unroll
      for (int r = 0; r < 8; ++r) {
        bf16x2 u = *reinterpret_cast<const bf16x2*>(Vp + 2 * r);
        bv[2 * r] = u[0]; bv[2 * r + 1] = u[1];
      }
      ot[t] = wmma_bf16(pa, bv, ot[t]);
    }
  }

  // normalize and store O as bf16 [B,S,D]
#pragma unroll
  for (int i = 0; i < 8; ++i) {
    const float inv = 1.f / fmaxf(l_run[i], 1e-20f);
    const int row = qbase + half * 8 + i;
    bf16_t* orow = o + ((size_t)(b * SEQ + row)) * DMODEL + h * DKV;
    orow[mcol]      = (bf16_t)(ot[0][i] * inv);
    orow[16 + mcol] = (bf16_t)(ot[1][i] * inv);
    orow[32 + mcol] = (bf16_t)(ot[2][i] * inv);
    orow[48 + mcol] = (bf16_t)(ot[3][i] * inv);
  }
}

// ---------------------------------------------------------------------------
// Host-side orchestration
// ---------------------------------------------------------------------------
extern "C" void kernel_launch(void* const* d_in, const int* in_sizes, int n_in,
                              void* d_out, int out_size, void* d_ws, size_t ws_size,
                              hipStream_t stream) {
  (void)in_sizes; (void)n_in; (void)out_size; (void)ws_size;
  const int*   ids  = (const int*)d_in[0];
  const int*   am   = (const int*)d_in[1];
  const float* emb  = (const float*)d_in[2];
  const float* pe   = (const float*)d_in[3];
  const float* wq   = (const float*)d_in[4];
  const float* wk   = (const float*)d_in[5];
  const float* wvw  = (const float*)d_in[6];
  const float* wo   = (const float*)d_in[7];
  const float* ln1w = (const float*)d_in[8];
  const float* ln1b = (const float*)d_in[9];
  const float* ln2w = (const float*)d_in[10];
  const float* ln2b = (const float*)d_in[11];
  const float* w1   = (const float*)d_in[12];
  const float* b1   = (const float*)d_in[13];
  const float* w2   = (const float*)d_in[14];
  const float* b2   = (const float*)d_in[15];
  const float* lnfw = (const float*)d_in[16];
  const float* lnfb = (const float*)d_in[17];

  char* ws = (char*)d_ws;
  size_t off = 0;
  auto alloc = [&](size_t bytes) -> char* {
    char* p = ws + off;
    off = (off + bytes + 255) & ~(size_t)255;
    return p;
  };
  float*  xf    = (float*) alloc((size_t)MTOK * DMODEL * 4);
  bf16_t* hbf   = (bf16_t*)alloc((size_t)MTOK * DMODEL * 2);
  bf16_t* qbf   = (bf16_t*)alloc((size_t)MTOK * DMODEL * 2);
  bf16_t* kbf   = (bf16_t*)alloc((size_t)MTOK * DMODEL * 2);
  bf16_t* vtb   = (bf16_t*)alloc((size_t)MTOK * DMODEL * 2);  // [B,H,DK,S]
  bf16_t* obf   = (bf16_t*)alloc((size_t)MTOK * DMODEL * 2);
  bf16_t* ffbf  = (bf16_t*)alloc((size_t)MTOK * FFDIM * 2);
  bf16_t* wqb   = (bf16_t*)alloc((size_t)DMODEL * DMODEL * 2);  // all [N,K]
  bf16_t* wkb   = (bf16_t*)alloc((size_t)DMODEL * DMODEL * 2);
  bf16_t* wvb   = (bf16_t*)alloc((size_t)DMODEL * DMODEL * 2);
  bf16_t* wob   = (bf16_t*)alloc((size_t)DMODEL * DMODEL * 2);
  bf16_t* w1b   = (bf16_t*)alloc((size_t)DMODEL * FFDIM * 2);  // [FF,D]
  bf16_t* w2b   = (bf16_t*)alloc((size_t)FFDIM * DMODEL * 2);  // [D,FF]
  bf16_t* embbf = (bf16_t*)alloc((size_t)VOCAB * DMODEL * 2);  // [V,D] natural

  auto cvtT = [&](const float* src, bf16_t* dst, int K, int N) {
    dim3 g((N + 31) / 32, (K + 31) / 32), blk(32, 8);
    k_cvt_transpose<<<g, blk, 0, stream>>>(src, dst, K, N);
  };

  {
    long long n = (long long)VOCAB * DMODEL;
    k_f32_to_bf16<<<32768, 256, 0, stream>>>(emb, embbf, n);
  }
  k_embed<<<MTOK, 256, 0, stream>>>(ids, emb, pe, xf);

  const dim3 gproj((DMODEL + 255) / 256, MTOK / 16);
  const dim3 gff1((FFDIM + 255) / 256, MTOK / 16);
  const dim3 gattn(BATCH * HNUM, SEQ / 64);

  for (int l = 0; l < LNUM; ++l) {
    cvtT(wq + (size_t)l * DMODEL * DMODEL, wqb, DMODEL, DMODEL);
    cvtT(wk + (size_t)l * DMODEL * DMODEL, wkb, DMODEL, DMODEL);
    cvtT(wvw + (size_t)l * DMODEL * DMODEL, wvb, DMODEL, DMODEL);
    cvtT(wo + (size_t)l * DMODEL * DMODEL, wob, DMODEL, DMODEL);
    cvtT(w1 + (size_t)l * DMODEL * FFDIM, w1b, DMODEL, FFDIM);
    cvtT(w2 + (size_t)l * FFDIM * DMODEL, w2b, FFDIM, DMODEL);

    k_layernorm_bf16<<<MTOK, 256, 0, stream>>>(xf, ln1w + (size_t)l * DMODEL,
                                               ln1b + (size_t)l * DMODEL, hbf);
    k_gemm_wmma<false, false, false, false, true, false><<<gproj, 128, 0, stream>>>(
        hbf, wqb, nullptr, nullptr, nullptr, qbf, MTOK, DMODEL, DMODEL);
    k_gemm_wmma<false, false, false, false, true, false><<<gproj, 128, 0, stream>>>(
        hbf, wkb, nullptr, nullptr, nullptr, kbf, MTOK, DMODEL, DMODEL);
    k_gemm_wmma<false, false, false, false, false, true><<<gproj, 128, 0, stream>>>(
        hbf, wvb, nullptr, nullptr, nullptr, vtb, MTOK, DMODEL, DMODEL);

    k_flash_attn<<<gattn, 128, 0, stream>>>(qbf, kbf, vtb, am, obf);

    k_gemm_wmma<false, true, false, true, false, false><<<gproj, 128, 0, stream>>>(
        obf, wob, nullptr, xf, xf, nullptr, MTOK, DMODEL, DMODEL);

    k_layernorm_bf16<<<MTOK, 256, 0, stream>>>(xf, ln2w + (size_t)l * DMODEL,
                                               ln2b + (size_t)l * DMODEL, hbf);
    k_gemm_wmma<true, false, true, false, true, false><<<gff1, 128, 0, stream>>>(
        hbf, w1b, b1 + (size_t)l * FFDIM, nullptr, nullptr, ffbf, MTOK, FFDIM, DMODEL);
    k_gemm_wmma<true, true, false, true, false, false><<<gproj, 128, 0, stream>>>(
        ffbf, w2b, b2 + (size_t)l * DMODEL, xf, xf, nullptr, MTOK, DMODEL, FFDIM);
  }

  k_layernorm_bf16<<<MTOK, 256, 0, stream>>>(xf, lnfw, lnfb, hbf);

  const dim3 glm((VOCAB + 255) / 256, MTOK / 16);
  k_gemm_wmma<false, false, false, true, false, false><<<glm, 128, 0, stream>>>(
      hbf, embbf, nullptr, nullptr, (float*)d_out, nullptr, MTOK, VOCAB, DMODEL);
}
